// GraphSAGESupplyChain_88373247083005
// MI455X (gfx1250) — compile-verified, matrix operand order checked by reading, not run on hardware
//
#include <hip/hip_runtime.h>
#include <hip/hip_bf16.h>

#define N_NODES 100000
#define N_EDGES 1600000
#define HDIM    128
#define LAYERS  3
#define BN_EPS  1e-5f
#define NORM_EPS 1e-12f

typedef __attribute__((ext_vector_type(16))) __bf16 v16bf;
typedef __attribute__((ext_vector_type(8)))  __bf16 v8bf;   // 16 B
typedef __attribute__((ext_vector_type(4)))  __bf16 v4bf;   // 8 B
typedef __attribute__((ext_vector_type(8)))  float  v8f;

// ---------------------------------------------------------------------------
// V_WMMA_F32_16X16X32_BF16 fragment layout (ISA 7.12.2, wave32):
// A 16x32: lane l<16 holds row m=l, K in {0..7, 16..23};
//          lane l>=16 holds row m=l-16, K in {8..15, 24..31}.
// B 32x16 mirrors with n in place of m; for out = A @ W^T (row-major W),
// B[k][n] = W[n][k] is k-contiguous, so the same loader serves both.
// ---------------------------------------------------------------------------

// fp32 source, optional per-lane scale folded into the conversion.
__device__ __forceinline__ v16bf frag_from_f32(const float* __restrict__ base,
                                               int ld, int lane, float scale) {
  const int half = (lane >> 4) << 3;
  const float* rp = base + (size_t)(lane & 15) * ld + half;
  v16bf f;
#pragma unroll
  for (int i = 0; i < 8; ++i) f[i] = (__bf16)(rp[i] * scale);
#pragma unroll
  for (int i = 0; i < 8; ++i) f[i + 8] = (__bf16)(rp[i + 16] * scale);
  return f;
}

// bf16 source: two straight b128 loads, zero VALU conversion.
__device__ __forceinline__ v16bf frag_from_bf16(const __bf16* __restrict__ base,
                                                int ld, int lane) {
  const int half = (lane >> 4) << 3;
  const __bf16* rp = base + (size_t)(lane & 15) * ld + half;
  const v8bf lo = *(const v8bf*)rp;          // K = half + 0..7
  const v8bf hi = *(const v8bf*)(rp + 16);   // K = half + 16..23
  v16bf f;
#pragma unroll
  for (int i = 0; i < 8; ++i) { f[i] = lo[i]; f[i + 8] = hi[i]; }
  return f;
}

// ---------------------------------------------------------------------------
// fp32 -> bf16 conversion (weights, one-time per launch)
// ---------------------------------------------------------------------------
__global__ void cvt_f32_bf16(const float* __restrict__ in,
                             __bf16* __restrict__ out, int n) {
  const int i = blockIdx.x * blockDim.x + threadIdx.x;
  if (i < n) out[i] = (__bf16)in[i];
}

// ---------------------------------------------------------------------------
// Encoder: hb = relu(x @ W^T + b), x fp32, W bf16, out bf16.
// grid.x = N/16, blockDim = 256 (8 waves = 8 col tiles of 16).
// ---------------------------------------------------------------------------
__global__ void encoder_gemm(const float* __restrict__ x,
                             const __bf16* __restrict__ W,
                             const float* __restrict__ bias,
                             __bf16* __restrict__ outb) {
  const int lane = threadIdx.x & 31;
  const int wave = threadIdx.x >> 5;
  const int row0 = blockIdx.x << 4;
  const int col0 = wave << 4;

  const float*  Ab = x + (size_t)row0 * HDIM;
  const __bf16* Wb = W + (size_t)col0 * HDIM;

  v8f c = {};
#pragma unroll
  for (int k0 = 0; k0 < HDIM; k0 += 32) {
    v16bf a = frag_from_f32(Ab + k0, HDIM, lane, 1.0f);
    v16bf b = frag_from_bf16(Wb + k0, HDIM, lane);
    c = __builtin_amdgcn_wmma_f32_16x16x32_bf16(false, a, false, b,
                                                (short)0, c, false, false);
  }
  const int n    = col0 + (lane & 15);
  const int moff = (lane >> 4) << 3;
  const float bv = bias[n];
#pragma unroll
  for (int r = 0; r < 8; ++r) {
    float v = c[r] + bv;
    v = v > 0.f ? v : 0.f;
    outb[(size_t)(row0 + moff + r) * HDIM + n] = (__bf16)v;
  }
}

// ---------------------------------------------------------------------------
// Head GEMM: out[M,ncols] = relu(hb[M,128] @ W[ncols,128]^T + bias), fp32 out.
// blockDim = (ncols/16)*32.
// ---------------------------------------------------------------------------
__global__ void head_gemm(const __bf16* __restrict__ A,
                          const __bf16* __restrict__ W,
                          const float* __restrict__ bias,
                          float* __restrict__ out, int ncols) {
  const int lane = threadIdx.x & 31;
  const int wave = threadIdx.x >> 5;
  const int row0 = blockIdx.x << 4;
  const int col0 = wave << 4;

  const __bf16* Ab = A + (size_t)row0 * HDIM;
  const __bf16* Wb = W + (size_t)col0 * HDIM;

  v8f c = {};
#pragma unroll
  for (int k0 = 0; k0 < HDIM; k0 += 32) {
    v16bf a = frag_from_bf16(Ab + k0, HDIM, lane);
    v16bf b = frag_from_bf16(Wb + k0, HDIM, lane);
    c = __builtin_amdgcn_wmma_f32_16x16x32_bf16(false, a, false, b,
                                                (short)0, c, false, false);
  }
  const int n    = col0 + (lane & 15);
  const int moff = (lane >> 4) << 3;
  const float bv = bias[n];
#pragma unroll
  for (int r = 0; r < 8; ++r) {
    float v = c[r] + bv;
    v = v > 0.f ? v : 0.f;
    out[(size_t)(row0 + moff + r) * ncols + n] = v;
  }
}

// ---------------------------------------------------------------------------
// Edge scatter: 32 lanes per edge, 8 B bf16 gather of h[src] (4 elems/lane),
// fp32 atomics into msg[dst] (precision kept in the accumulator), degree
// atomic, and a global_prefetch of the source row 2048 edges ahead.
// ---------------------------------------------------------------------------
__global__ void edge_scatter(const int* __restrict__ src,
                             const int* __restrict__ dst,
                             const __bf16* __restrict__ h,
                             float* __restrict__ msg,
                             float* __restrict__ deg, int nEdges) {
  const int gid  = blockIdx.x * blockDim.x + threadIdx.x;
  const int e    = gid >> 5;
  const int lane = gid & 31;
  if (e >= nEdges) return;

  const int e2 = e + 2048;
  if (e2 < nEdges) {
    const int s2 = src[e2];
    __builtin_prefetch(h + (size_t)s2 * HDIM + lane * 4, 0, 1);
  }

  const int s = src[e];
  const int d = dst[e];
  const v4bf hv = *(const v4bf*)(h + (size_t)s * HDIM + lane * 4);
  float* mp = msg + (size_t)d * HDIM + lane * 4;
  atomicAdd(mp + 0, (float)hv[0]);
  atomicAdd(mp + 1, (float)hv[1]);
  atomicAdd(mp + 2, (float)hv[2]);
  atomicAdd(mp + 3, (float)hv[3]);
  if (lane == 0) atomicAdd(deg + d, 1.0f);
}

// ---------------------------------------------------------------------------
// Fused SAGE combine:
//   out = (msg/deg) @ Wl^T + bl + h @ Wr^T   (dual WMMA accumulate)
//   out /= max(||out||_2, 1e-12)             (shuffle + LDS row reduce)
//   out = relu(BN_eval(out))
// Writes bf16 h for the next layer; on the last layer also writes fp32 h
// straight into the output tuple slot (houtf != nullptr).
// ---------------------------------------------------------------------------
__global__ void sage_combine(const __bf16* __restrict__ hin,
                             const float* __restrict__ msg,
                             const float* __restrict__ deg,
                             const __bf16* __restrict__ wl,
                             const float* __restrict__ bl,
                             const __bf16* __restrict__ wr,
                             const float* __restrict__ gamma,
                             const float* __restrict__ beta,
                             const float* __restrict__ mean,
                             const float* __restrict__ var,
                             __bf16* __restrict__ houtb,
                             float* __restrict__ houtf) {
  __shared__ float rowssq[16];
  const int lane = threadIdx.x & 31;
  const int wave = threadIdx.x >> 5;
  const int row0 = blockIdx.x << 4;
  const int col0 = wave << 4;

  if (threadIdx.x < 16) rowssq[threadIdx.x] = 0.f;
  __syncthreads();

  const float dv   = deg[row0 + (lane & 15)];
  const float invd = 1.0f / fmaxf(dv, 1.0f);

  const float*  Mb = msg + (size_t)row0 * HDIM;
  const __bf16* Hb = hin + (size_t)row0 * HDIM;
  const __bf16* Wl = wl + (size_t)col0 * HDIM;
  const __bf16* Wr = wr + (size_t)col0 * HDIM;

  v8f c = {};
#pragma unroll
  for (int k0 = 0; k0 < HDIM; k0 += 32) {
    v16bf am = frag_from_f32(Mb + k0, HDIM, lane, invd);
    v16bf bw = frag_from_bf16(Wl + k0, HDIM, lane);
    c = __builtin_amdgcn_wmma_f32_16x16x32_bf16(false, am, false, bw,
                                                (short)0, c, false, false);
    v16bf ah = frag_from_bf16(Hb + k0, HDIM, lane);
    v16bf br = frag_from_bf16(Wr + k0, HDIM, lane);
    c = __builtin_amdgcn_wmma_f32_16x16x32_bf16(false, ah, false, br,
                                                (short)0, c, false, false);
  }

  const int n    = col0 + (lane & 15);
  const int moff = (lane >> 4) << 3;
  const float blv = bl[n];

  float v[8];
  float sq[8];
#pragma unroll
  for (int r = 0; r < 8; ++r) {
    v[r] = c[r] + blv;
    float s = v[r] * v[r];
#pragma unroll
    for (int off = 1; off < 16; off <<= 1) s += __shfl_xor(s, off, 32);
    sq[r] = s;
  }
  if ((lane & 15) == 0) {
#pragma unroll
    for (int r = 0; r < 8; ++r) atomicAdd(&rowssq[moff + r], sq[r]);
  }
  __syncthreads();

  const float scl = gamma[n] * rsqrtf(var[n] + BN_EPS);
  const float sft = beta[n] - mean[n] * scl;
#pragma unroll
  for (int r = 0; r < 8; ++r) {
    const float rn = 1.0f / fmaxf(sqrtf(rowssq[moff + r]), NORM_EPS);
    float o = v[r] * rn;
    o = o * scl + sft;
    o = o > 0.f ? o : 0.f;
    const size_t idx = (size_t)(row0 + moff + r) * HDIM + n;
    houtb[idx] = (__bf16)o;
    if (houtf) houtf[idx] = o;
  }
}

// ---------------------------------------------------------------------------
// Final head projection: out[row] = dot(hid[row,:K], w2[:K]) + b2
// one wave per row, shuffle reduce.
// ---------------------------------------------------------------------------
__global__ void head_proj(const float* __restrict__ hid,
                          const float* __restrict__ w2,
                          const float* __restrict__ b2,
                          float* __restrict__ out, int K) {
  const int row  = blockIdx.x * (blockDim.x >> 5) + (threadIdx.x >> 5);
  const int lane = threadIdx.x & 31;
  if (row >= N_NODES) return;
  float s = 0.f;
  for (int k = lane; k < K; k += 32) s += hid[(size_t)row * K + k] * w2[k];
#pragma unroll
  for (int off = 16; off > 0; off >>= 1) s += __shfl_xor(s, off, 32);
  if (lane == 0) out[row] = s + b2[0];
}

// ---------------------------------------------------------------------------
extern "C" void kernel_launch(void* const* d_in, const int* in_sizes, int n_in,
                              void* d_out, int out_size, void* d_ws, size_t ws_size,
                              hipStream_t stream) {
  const float* x       = (const float*)d_in[0];
  const int*   ei      = (const int*)d_in[1];
  const float* enc_w   = (const float*)d_in[2];
  const float* enc_b   = (const float*)d_in[3];
  const float* lin_l_w = (const float*)d_in[4];
  const float* lin_l_b = (const float*)d_in[5];
  const float* lin_r_w = (const float*)d_in[6];
  const float* bn_g    = (const float*)d_in[7];
  const float* bn_b    = (const float*)d_in[8];
  const float* bn_m    = (const float*)d_in[9];
  const float* bn_v    = (const float*)d_in[10];
  const float* out_w1  = (const float*)d_in[11];
  const float* out_b1  = (const float*)d_in[12];
  const float* out_w2  = (const float*)d_in[13];
  const float* out_b2  = (const float*)d_in[14];
  const float* cost_w1 = (const float*)d_in[15];
  const float* cost_b1 = (const float*)d_in[16];
  const float* cost_w2 = (const float*)d_in[17];
  const float* cost_b2 = (const float*)d_in[18];
  const float* bw_w1   = (const float*)d_in[19];
  const float* bw_b1   = (const float*)d_in[20];
  const float* bw_w2   = (const float*)d_in[21];
  const float* bw_b2   = (const float*)d_in[22];

  const int* src = ei;
  const int* dst = ei + N_EDGES;

  const size_t NH = (size_t)N_NODES * HDIM;

  // ---- workspace layout (all offsets 128 B aligned) ----
  char* wsp = (char*)d_ws;
  __bf16* hbA = (__bf16*)wsp;                 wsp += NH * sizeof(__bf16);
  __bf16* hbB = (__bf16*)wsp;                 wsp += NH * sizeof(__bf16);
  float*  msg = (float*)wsp;                  wsp += NH * sizeof(float);
  float*  deg = (float*)wsp;                  wsp += (size_t)N_NODES * sizeof(float) + 256;
  __bf16* enc_wb  = (__bf16*)wsp;             wsp += 16384 * sizeof(__bf16);
  __bf16* lwl_b   = (__bf16*)wsp;             wsp += 3 * 16384 * sizeof(__bf16);
  __bf16* lwr_b   = (__bf16*)wsp;             wsp += 3 * 16384 * sizeof(__bf16);
  __bf16* out_w1b = (__bf16*)wsp;             wsp += 16384 * sizeof(__bf16);
  __bf16* cost_w1b = (__bf16*)wsp;            wsp += 8192 * sizeof(__bf16);
  __bf16* bw_w1b  = (__bf16*)wsp;             wsp += 8192 * sizeof(__bf16);

  float* order_out = (float*)d_out;
  float* cost_out  = order_out + N_NODES;
  float* bw_out    = cost_out + N_NODES;
  float* h_out     = bw_out + N_NODES;

  const int rowTiles = N_NODES / 16;                  // 6250
  const int scatterBlocks = (N_EDGES * 32) / 256;     // 200000
  const int projBlocks = (N_NODES + 7) / 8;           // 12500

  // ---- one-shot weight conversion to bf16 ----
  cvt_f32_bf16<<<64, 256, 0, stream>>>(enc_w, enc_wb, 16384);
  cvt_f32_bf16<<<192, 256, 0, stream>>>(lin_l_w, lwl_b, 3 * 16384);
  cvt_f32_bf16<<<192, 256, 0, stream>>>(lin_r_w, lwr_b, 3 * 16384);
  cvt_f32_bf16<<<64, 256, 0, stream>>>(out_w1, out_w1b, 16384);
  cvt_f32_bf16<<<32, 256, 0, stream>>>(cost_w1, cost_w1b, 8192);
  cvt_f32_bf16<<<32, 256, 0, stream>>>(bw_w1, bw_w1b, 8192);

  // ---- encoder ----
  encoder_gemm<<<rowTiles, 256, 0, stream>>>(x, enc_wb, enc_b, hbA);

  // ---- SAGE layers ----
  __bf16* hin = hbA;
  __bf16* hnx = hbB;
  for (int i = 0; i < LAYERS; ++i) {
    hipMemsetAsync(msg, 0, NH * sizeof(float), stream);
    hipMemsetAsync(deg, 0, (size_t)N_NODES * sizeof(float), stream);
    edge_scatter<<<scatterBlocks, 256, 0, stream>>>(src, dst, hin, msg, deg, N_EDGES);
    sage_combine<<<rowTiles, 256, 0, stream>>>(
        hin, msg, deg,
        lwl_b + (size_t)i * 16384, lin_l_b + (size_t)i * HDIM,
        lwr_b + (size_t)i * 16384,
        bn_g + (size_t)i * HDIM, bn_b + (size_t)i * HDIM,
        bn_m + (size_t)i * HDIM, bn_v + (size_t)i * HDIM,
        hnx, (i == LAYERS - 1) ? h_out : (float*)nullptr);
    __bf16* t = hin; hin = hnx; hnx = t;
  }
  // hin = final bf16 embeddings; fp32 copy already in d_out.

  // ---- heads (hidden buffer reuses msg) ----
  head_gemm<<<rowTiles, 256, 0, stream>>>(hin, out_w1b, out_b1, msg, HDIM);
  head_proj<<<projBlocks, 256, 0, stream>>>(msg, out_w2, out_b2, order_out, HDIM);

  head_gemm<<<rowTiles, 128, 0, stream>>>(hin, cost_w1b, cost_b1, msg, 64);
  head_proj<<<projBlocks, 256, 0, stream>>>(msg, cost_w2, cost_b2, cost_out, 64);

  head_gemm<<<rowTiles, 128, 0, stream>>>(hin, bw_w1b, bw_b1, msg, 64);
  head_proj<<<projBlocks, 256, 0, stream>>>(msg, bw_w2, bw_b2, bw_out, 64);
}
